// FlaxBeitSelfAttention_55336358642293
// MI455X (gfx1250) — compile-verified, matrix operand order checked by reading, not run on hardware
//
#include <hip/hip_runtime.h>
#include <hip/hip_bf16.h>

#define HEADS  12
#define HD     64
#define SEQ    197
#define BATCH  64
#define HIDDEN 768
#define MTOT   (BATCH * SEQ)   // 12608
#define SPAD   208             // 13*16 (score tile span)
#define SPAD2  224             // 7*32  (padded K-dim for P@V)
#define PSTRIDE 232            // LDS prob row stride (bank-conflict free, 16B aligned)

typedef __attribute__((ext_vector_type(16))) __bf16 v16bf;
typedef __attribute__((ext_vector_type(8)))  __bf16 v8bf;
typedef __attribute__((ext_vector_type(4)))  __bf16 v4bf;
typedef __attribute__((ext_vector_type(8)))  float  v8f;
typedef __attribute__((ext_vector_type(4)))  float  f4;

static __device__ __forceinline__ v8f vzero8() {
  v8f v = {0.f,0.f,0.f,0.f,0.f,0.f,0.f,0.f};
  return v;
}

// Load a 16x32 bf16 A/B fragment's per-lane 16 elements.
// p points at (row base + kstep + half*8); elements 0..7 from p, 8..15 from p+16.
static __device__ __forceinline__ v16bf frag16(const __bf16* p) {
  v16bf r;
  ((v8bf*)&r)[0] = ((const v8bf*)p)[0];
  ((v8bf*)&r)[1] = ((const v8bf*)(p + 16))[0];
  return r;
}

static __device__ __forceinline__ v8f wmma_bf16(v16bf a, v16bf b, v8f c) {
  // (neg_a, A, neg_b, B, c_mod, C, reuse_a, reuse_b)
  return __builtin_amdgcn_wmma_f32_16x16x32_bf16(false, a, false, b, (short)0, c,
                                                 false, false);
}

// CDNA5 async global->LDS 16B copy (no VGPR round trip, tracked by ASYNCcnt).
#define ASYNC_LD16(ldsoff, gptr)                                            \
  asm volatile("global_load_async_to_lds_b128 %0, %1, off"                  \
               :: "v"(ldsoff), "v"(gptr) : "memory")

// ---------------------------------------------------------------------------
// Kernel 0a: X f32 -> bf16 (vectorized, exact coverage: 12608*768 % 1024 == 0)
// ---------------------------------------------------------------------------
__global__ void xconv(const float* __restrict__ X, __bf16* __restrict__ Xb) {
  size_t idx = ((size_t)blockIdx.x * 256 + threadIdx.x) * 4;
  f4 v = *(const f4*)(X + idx);
  v4bf o;
#pragma unroll
  for (int e = 0; e < 4; e++) o[e] = (__bf16)v[e];
  *(v4bf*)(Xb + idx) = o;
}

// ---------------------------------------------------------------------------
// Kernel 0b: weight transpose+convert: Wt[z][n][k] = (bf16)W_z[k][n]
// 32x32 LDS tiles, coalesced reads and writes.
// ---------------------------------------------------------------------------
__global__ void wtrans(const float* __restrict__ Wq, const float* __restrict__ Wk,
                       const float* __restrict__ Wv, __bf16* __restrict__ Wt) {
  __shared__ __align__(16) float t[32][33];
  const int z = blockIdx.z;
  const float* W = (z == 0) ? Wq : (z == 1) ? Wk : Wv;
  const int k0 = blockIdx.x * 32, n0 = blockIdx.y * 32;
  const int tx = threadIdx.x & 31, ty = threadIdx.x >> 5;   // 32 x 8
#pragma unroll
  for (int i = 0; i < 32; i += 8)
    t[ty + i][tx] = W[(size_t)(k0 + ty + i) * HIDDEN + n0 + tx];
  __syncthreads();
#pragma unroll
  for (int i = 0; i < 32; i += 8)
    Wt[((size_t)z * HIDDEN + n0 + ty + i) * HIDDEN + k0 + tx] = (__bf16)t[tx][ty + i];
}

// ---------------------------------------------------------------------------
// Kernel 1: relative-position bias [12][208][208] f32, -inf in padding
// (padding acts as the softmax mask for k >= 197 / q >= 197).
// ---------------------------------------------------------------------------
__global__ void bias_build(const float* __restrict__ table, float* __restrict__ bias) {
  int idx = blockIdx.x * blockDim.x + threadIdx.x;
  if (idx >= HEADS * SPAD * SPAD) return;
  int k = idx % SPAD;
  int q = (idx / SPAD) % SPAD;
  int h = idx / (SPAD * SPAD);
  float v;
  if (q >= SEQ || k >= SEQ) {
    v = -__builtin_inff();
  } else {
    int r;
    if (q == 0 && k == 0)      r = 731;
    else if (k == 0)           r = 730;
    else if (q == 0)           r = 729;
    else {
      int qh = (q - 1) / 14, qw = (q - 1) % 14;
      int kh = (k - 1) / 14, kw = (k - 1) % 14;
      r = (qh - kh + 13) * 27 + (qw - kw + 13);
    }
    v = table[r * HEADS + h];
  }
  bias[idx] = v;
}

// ---------------------------------------------------------------------------
// Kernel 2: QKV projection GEMM (bf16 WMMA, f32 accumulate), async double-
// buffered global->LDS staging.
// grid = (N/64, ceil(M/128), 3); block = 256 (8 waves, 4x2 of 32x32 tiles).
// z=0: Q -> (acc+bq)*0.125 as [bh][s][64] bf16
// z=1: K ->  acc           as [bh][s][64] bf16
// z=2: V -> (acc+bv)       as [bh][d][224] bf16 (pre-transposed for P@V)
// ---------------------------------------------------------------------------
__global__ __launch_bounds__(256) void qkv_gemm(
    const __bf16* __restrict__ Xb, const __bf16* __restrict__ Wt,
    const float* __restrict__ Bq, const float* __restrict__ Bv,
    __bf16* __restrict__ Qb, __bf16* __restrict__ Kb, __bf16* __restrict__ Vt)
{
  __shared__ __align__(16) __bf16 As[2][128][40];  // [m][k], stride 40: conflict-free
  __shared__ __align__(16) __bf16 Bs[2][64][40];   // [n][k] (pre-transposed weights)

  const int z  = blockIdx.z;
  const int m0 = blockIdx.y * 128;
  const int n0 = blockIdx.x * 64;
  const int tid  = threadIdx.x;
  const int wave = tid >> 5, lane = tid & 31;
  const int wm = wave & 3, wn = wave >> 2;
  const int half = lane >> 4, lm = lane & 15;

  v8f acc[2][2];
#pragma unroll
  for (int i = 0; i < 2; i++)
#pragma unroll
    for (int j = 0; j < 2; j++) acc[i][j] = vzero8();

  // ---- async staging setup: 3 x 16B transfers per thread per k-step ----
  const int row = tid >> 2;          // 0..63
  const int ac  = (tid & 3) * 8;     // chunk element offset (0/8/16/24)
  const unsigned asb = (unsigned)(unsigned long long)(&As[0][0][0]);
  const unsigned bsb = (unsigned)(unsigned long long)(&Bs[0][0][0]);
  const unsigned abytes = (unsigned)sizeof(As[0]);   // 10240
  const unsigned bbytes = (unsigned)sizeof(Bs[0]);   // 5120
  const unsigned ldsA0 = asb + (unsigned)(row * 80 + ac * 2);
  const unsigned ldsA1 = asb + (unsigned)((row + 64) * 80 + ac * 2);
  const unsigned ldsB  = bsb + (unsigned)(row * 80 + ac * 2);
  // Rows m >= MTOT over-read into the Wt buffer (finite bf16); those D rows
  // are never stored (WMMA rows are independent).
  const __bf16* gA0 = Xb + (size_t)(m0 + row) * HIDDEN + ac;
  const __bf16* gA1 = Xb + (size_t)(m0 + row + 64) * HIDDEN + ac;
  const __bf16* gB  = Wt + ((size_t)z * HIDDEN + n0 + row) * HIDDEN + ac;

  // prologue: issue tile kk=0 into buffer 0
  ASYNC_LD16(ldsA0, gA0);
  ASYNC_LD16(ldsA1, gA1);
  ASYNC_LD16(ldsB,  gB);

  int buf = 0;
  for (int kk = 0; kk < HIDDEN; kk += 32) {
    if (kk + 32 < HIDDEN) {
      const unsigned nb = (unsigned)(buf ^ 1);
      ASYNC_LD16(ldsA0 + nb * abytes, gA0 + kk + 32);
      ASYNC_LD16(ldsA1 + nb * abytes, gA1 + kk + 32);
      ASYNC_LD16(ldsB  + nb * bbytes, gB  + kk + 32);
      // 6 outstanding; <=3 retires exactly the previous group (in-order).
      asm volatile("s_wait_asynccnt 0x3" ::: "memory");
    } else {
      asm volatile("s_wait_asynccnt 0x0" ::: "memory");
    }
    __syncthreads();

    // ---- 4 WMMAs per wave per k-step ----
    v16bf a0 = frag16(&As[buf][wm * 32 + lm][half * 8]);
    v16bf a1 = frag16(&As[buf][wm * 32 + 16 + lm][half * 8]);
    v16bf b0 = frag16(&Bs[buf][wn * 32 + lm][half * 8]);
    v16bf b1 = frag16(&Bs[buf][wn * 32 + 16 + lm][half * 8]);
    acc[0][0] = wmma_bf16(a0, b0, acc[0][0]);
    acc[0][1] = wmma_bf16(a0, b1, acc[0][1]);
    acc[1][0] = wmma_bf16(a1, b0, acc[1][0]);
    acc[1][1] = wmma_bf16(a1, b1, acc[1][1]);
    __syncthreads();
    buf ^= 1;
  }

  // ---- epilogue: bias/scale, bf16 convert, scatter to staging layouts ----
#pragma unroll
  for (int i = 0; i < 2; i++) {
#pragma unroll
    for (int r = 0; r < 8; r++) {
      int gm = m0 + wm * 32 + i * 16 + r + 8 * half;
      if (gm >= MTOT) continue;
      int b = gm / SEQ, s = gm - b * SEQ;
#pragma unroll
      for (int j = 0; j < 2; j++) {
        int gn = n0 + wn * 32 + j * 16 + lm;
        int hh = gn >> 6, d = gn & 63;
        int bh = b * HEADS + hh;
        float v = acc[i][j][r];
        if (z == 0) {
          v = (v + Bq[gn]) * 0.125f;  // 1/sqrt(64)
          Qb[((size_t)bh * SEQ + s) * HD + d] = (__bf16)v;
        } else if (z == 1) {
          Kb[((size_t)bh * SEQ + s) * HD + d] = (__bf16)v;
        } else {
          v = v + Bv[gn];
          Vt[((size_t)bh * HD + d) * SPAD2 + s] = (__bf16)v;
        }
      }
    }
  }
}

// ---------------------------------------------------------------------------
// Kernel 3: fused attention per (b,h). 256 threads; each wave owns a 16-row
// query tile: QK^T (WMMA) + bias -> register softmax (shfl over lane halves)
// -> bf16 probs via per-wave LDS relayout -> P@V (WMMA) -> f32 out.
// ---------------------------------------------------------------------------
__global__ __launch_bounds__(256, 1) void attn(
    const __bf16* __restrict__ Qb, const __bf16* __restrict__ Kb,
    const __bf16* __restrict__ Vt, const float* __restrict__ bias,
    float* __restrict__ out)
{
  __shared__ __align__(16) __bf16 P[8][16][PSTRIDE];   // per-wave prob tiles

  const int bh = blockIdx.x;
  const int b = bh / HEADS, h = bh % HEADS;
  const int tid  = threadIdx.x;
  const int wave = tid >> 5, lane = tid & 31;
  const int half = lane >> 4, lm = lane & 15;

  const __bf16* Qh = Qb + (size_t)bh * SEQ * HD;
  const __bf16* Kh = Kb + (size_t)bh * SEQ * HD;
  const __bf16* Vh = Vt + (size_t)bh * HD * SPAD2;
  const float*  Bh = bias + (size_t)h * SPAD * SPAD;

  for (int qt = wave; qt < 13; qt += 8) {
    // Q fragments for d-steps 0 and 32 (kept in regs across all k-tiles)
    const __bf16* qrow = Qh + (size_t)(qt * 16 + lm) * HD;
    v16bf qf0 = frag16(qrow + half * 8);
    v16bf qf1 = frag16(qrow + 32 + half * 8);

    // ---- scores: 13 16x16 tiles, 2 WMMAs each ----
    v8f sc[13];
#pragma unroll
    for (int kt = 0; kt < 13; kt++) {
      const __bf16* krow = Kh + (size_t)(kt * 16 + lm) * HD;
      v16bf kf0 = frag16(krow + half * 8);
      v16bf kf1 = frag16(krow + 32 + half * 8);
      v8f c = vzero8();
      c = wmma_bf16(qf0, kf0, c);
      c = wmma_bf16(qf1, kf1, c);
      const float* bp = Bh + (size_t)(qt * 16 + 8 * half) * SPAD + kt * 16 + lm;
#pragma unroll
      for (int r = 0; r < 8; r++) c[r] += bp[r * SPAD];
      sc[kt] = c;
    }

    // ---- row softmax: row m lives in VGPR r of one 16-lane half ----
    v8f mx = sc[0];
#pragma unroll
    for (int kt = 1; kt < 13; kt++)
#pragma unroll
      for (int r = 0; r < 8; r++) mx[r] = fmaxf(mx[r], sc[kt][r]);
#pragma unroll
    for (int off = 1; off < 16; off <<= 1)
#pragma unroll
      for (int r = 0; r < 8; r++) mx[r] = fmaxf(mx[r], __shfl_xor(mx[r], off, 32));

    v8f sum = vzero8();
#pragma unroll
    for (int kt = 0; kt < 13; kt++)
#pragma unroll
      for (int r = 0; r < 8; r++) {
        float e = __expf(sc[kt][r] - mx[r]);
        sc[kt][r] = e;
        sum[r] += e;
      }
#pragma unroll
    for (int off = 1; off < 16; off <<= 1)
#pragma unroll
      for (int r = 0; r < 8; r++) sum[r] += __shfl_xor(sum[r], off, 32);

    float inv[8];
#pragma unroll
    for (int r = 0; r < 8; r++) inv[r] = 1.0f / sum[r];

    // ---- probs -> bf16 in LDS (C-layout store, A-layout reload) ----
#pragma unroll
    for (int kt = 0; kt < 13; kt++)
#pragma unroll
      for (int r = 0; r < 8; r++)
        P[wave][r + 8 * half][kt * 16 + lm] = (__bf16)(sc[kt][r] * inv[r]);
#pragma unroll
    for (int r = 0; r < 8; r++)            // zero cols 208..223
      P[wave][r + 8 * half][208 + lm] = (__bf16)0.0f;

    // ---- P @ V: 7 k-steps x 4 d-tiles ----
    v8f o[4];
#pragma unroll
    for (int nt = 0; nt < 4; nt++) o[nt] = vzero8();
#pragma unroll
    for (int ks = 0; ks < 7; ks++) {
      v16bf pa = frag16(&P[wave][lm][ks * 32 + half * 8]);
#pragma unroll
      for (int nt = 0; nt < 4; nt++) {
        v16bf vb = frag16(Vh + (size_t)(nt * 16 + lm) * SPAD2 + ks * 32 + half * 8);
        o[nt] = wmma_bf16(pa, vb, o[nt]);
      }
    }

    // ---- store f32 output [b][s][h*64+d] ----
#pragma unroll
    for (int r = 0; r < 8; r++) {
      int s = qt * 16 + r + 8 * half;
      if (s >= SEQ) continue;
      float* op = out + ((size_t)b * SEQ + s) * HIDDEN + h * HD;
#pragma unroll
      for (int nt = 0; nt < 4; nt++) op[nt * 16 + lm] = o[nt][r];
    }
  }
}

// ---------------------------------------------------------------------------
extern "C" void kernel_launch(void* const* d_in, const int* in_sizes, int n_in,
                              void* d_out, int out_size, void* d_ws, size_t ws_size,
                              hipStream_t stream) {
  (void)in_sizes; (void)n_in; (void)out_size; (void)ws_size;
  const float* X  = (const float*)d_in[0];
  const float* Wq = (const float*)d_in[1];
  const float* Bq = (const float*)d_in[2];
  const float* Wk = (const float*)d_in[3];
  const float* Wv = (const float*)d_in[4];
  const float* Bv = (const float*)d_in[5];
  const float* tb = (const float*)d_in[6];

  // Workspace layout (order matters: deliberate over-reads must land on
  // finite bf16 data): Xb | Wt | Qb | Kb | Vt | bias
  char* ws = (char*)d_ws;
  const size_t XBbytes = (size_t)MTOT * HIDDEN * 2;                  // 19,365,888
  const size_t WTbytes = (size_t)3 * HIDDEN * HIDDEN * 2;            //  3,538,944
  const size_t QKbytes = (size_t)BATCH * HEADS * SEQ * HD * 2;       // 19,365,888
  const size_t VTbytes = (size_t)BATCH * HEADS * HD * SPAD2 * 2;     // 22,020,096
  __bf16* Xb   = (__bf16*)ws;
  __bf16* Wt   = (__bf16*)(ws + XBbytes);
  __bf16* Qb   = (__bf16*)(ws + XBbytes + WTbytes);
  __bf16* Kb   = (__bf16*)(ws + XBbytes + WTbytes + QKbytes);
  __bf16* Vt   = (__bf16*)(ws + XBbytes + WTbytes + 2 * QKbytes);
  float*  bias = (float*)(ws + XBbytes + WTbytes + 2 * QKbytes + VTbytes);

  xconv<<<dim3((MTOT * HIDDEN) / 1024), dim3(256), 0, stream>>>(X, Xb);
  wtrans<<<dim3(HIDDEN / 32, HIDDEN / 32, 3), dim3(256), 0, stream>>>(Wq, Wk, Wv, Wt);
  bias_build<<<dim3((HEADS * SPAD * SPAD + 255) / 256), dim3(256), 0, stream>>>(tb, bias);
  qkv_gemm<<<dim3(HIDDEN / 64, (MTOT + 127) / 128, 3), dim3(256), 0, stream>>>(
      Xb, Wt, Bq, Bv, Qb, Kb, Vt);
  attn<<<dim3(BATCH * HEADS), dim3(256), 0, stream>>>(Qb, Kb, Vt, bias, (float*)d_out);
}
